// _QuantumGate_65481071404536
// MI455X (gfx1250) — matrix operand order repaired and checked by reading, not compiled, not run on hardware
//
#include <hip/hip_runtime.h>
#include <math.h>

typedef __attribute__((ext_vector_type(16))) _Float16     v16h;
typedef __attribute__((ext_vector_type(8)))  _Float16     v8h;
typedef __attribute__((ext_vector_type(8)))  float        v8f;
typedef __attribute__((ext_vector_type(4)))  unsigned int u32x4;
typedef __attribute__((ext_vector_type(8)))  int          i32x8;
typedef __attribute__((ext_vector_type(4)))  int          i32x4;

#define QN_WIRES 10
#define QDIM     1024
#define QDEPTH   2

// ---------------------------------------------------------------------------
// Single-workgroup precompute: simulate U_fixed on the 16 basis states
// |k> (x) |0^6> entirely in LDS (16 x 1024 complex f32 = 128KB), then emit the
// packed GEMM operands P||Q (2048 rows x 32 f16 = 128KB, contiguous) to ws.
//   P[i, k]   =  Mr[i,k]   P[i,16+k] = -Mi[i,k]     (Re(st) = F . P-row)
//   Q[i, k]   =  Mi[i,k]   Q[i,16+k] =  Mr[i,k]     (Im(st) = F . Q-row)
// ---------------------------------------------------------------------------
__global__ void __launch_bounds__(1024)
k_precompute(const float* __restrict__ theta, _Float16* __restrict__ PQ) {
    extern __shared__ float LS[];          // 16*1024*2 floats = 128KB
    int tid = threadIdx.x;                 // 0..1023

    // init: column k = basis state index (k<<6)
#pragma unroll
    for (int k = 0; k < 16; ++k) {
        LS[2 * (k * QDIM + tid)]     = (tid == (k << 6)) ? 1.0f : 0.0f;
        LS[2 * (k * QDIM + tid) + 1] = 0.0f;
    }
    __syncthreads();

    for (int d = 0; d < QDEPTH; ++d) {
        for (int w = 0; w < QN_WIRES; ++w) {
            float h = 0.5f * theta[w];
            float c = cosf(h), s = sinf(h);
            int bit = 1 << (9 - w);
            int lowmask = bit - 1;
            // 16 cols x 512 pairs = 8192 butterflies, 8 per thread, disjoint
#pragma unroll
            for (int r = 0; r < 8; ++r) {
                int pid = (r << 10) | tid;
                int k = pid >> 9, p = pid & 511;
                int i0 = ((p & ~lowmask) << 1) | (p & lowmask);
                int i1 = i0 | bit;
                float* a0 = LS + 2 * (k * QDIM + i0);
                float* a1 = LS + 2 * (k * QDIM + i1);
                float a0r = a0[0], a0i = a0[1], a1r = a1[0], a1i = a1[1];
                // n0 = c*a0 - i*s*a1 ; n1 = -i*s*a0 + c*a1
                a0[0] = c * a0r + s * a1i;
                a0[1] = c * a0i - s * a1r;
                a1[0] = s * a0i + c * a1r;
                a1[1] = c * a1i - s * a0r;
            }
            __syncthreads();
        }
        // ring-CNOT perm: new[:, i] = old[:, RING_PERM[i]],
        // RING_PERM[i] = C_0(C_1(...C_8(C_ring(i))...))
        int i = tid;
        int j = i ^ ((i & 1) << 9);                    // CNOT(c=9,t=0)
#pragma unroll
        for (int w = 8; w >= 0; --w)                   // CNOT(c=w,t=w+1)
            j ^= (((j >> (9 - w)) & 1) << (8 - w));
        float tr[16], ti[16];
#pragma unroll
        for (int k = 0; k < 16; ++k) {
            tr[k] = LS[2 * (k * QDIM + j)];
            ti[k] = LS[2 * (k * QDIM + j) + 1];
        }
        __syncthreads();
#pragma unroll
        for (int k = 0; k < 16; ++k) {
            LS[2 * (k * QDIM + i)]     = tr[k];
            LS[2 * (k * QDIM + i) + 1] = ti[k];
        }
        __syncthreads();
    }

    // pack P (rows 0..1023) and Q (rows 1024..2047), 32 f16 per row
    int i = tid;
#pragma unroll
    for (int k = 0; k < 16; ++k) {
        float re = LS[2 * (k * QDIM + i)];
        float im = LS[2 * (k * QDIM + i) + 1];
        PQ[i * 32 + k]                   = (_Float16)re;
        PQ[i * 32 + 16 + k]              = (_Float16)(-im);
        PQ[(QDIM + i) * 32 + k]          = (_Float16)im;
        PQ[(QDIM + i) * 32 + 16 + k]     = (_Float16)re;
    }
}

// ---------------------------------------------------------------------------
// Fused evaluation. Per 256-thread block: 8 waves x 16-batch tiles.
//  - wave 0 issues ONE Tensor-Data-Mover load of P||Q (2048x32 f16, 128KB)
//    into LDS, completion via s_wait_tensorcnt, published by the barrier
//  - threads 0..127 compute their batch row's 32 real features into LDS
//  - each wave runs 64 chunks: Re/Im 16x16 via v_wmma_f32_16x16x32_f16
//    (A = feature tile from LDS, B = P/Q chunk from LDS), squares, and
//    accumulates 10 Z-sign sums, finally reduced with shfl_xor butterflies.
// ---------------------------------------------------------------------------
__global__ void __launch_bounds__(256)
k_eval(const float* __restrict__ x, const _Float16* __restrict__ PQg,
       float* __restrict__ out) {
    extern __shared__ char smem[];
    _Float16* feat = (_Float16*)smem;            // 128 rows x 32 f16 = 8KB
    _Float16* PQ   = (_Float16*)(smem + 8192);   // 2048 rows x 32 f16 = 128KB

    int tid  = threadIdx.x;
    int wave = tid >> 5;
    int lane = tid & 31;
    int nrow = lane & 15;

    if (wave == 0) {
        // TDM descriptor (D#): 2D tensor, 2048 rows x 32 halves, full tile,
        // data_size=2B, dim0_stride=32 elements -> contiguous 128KB copy.
        unsigned long long ga = (unsigned long long)(uintptr_t)PQg;
        u32x4 g0 = { 1u,                                   // count=1 (valid D#)
                     8192u,                                // lds_addr (bytes)
                     (unsigned)(ga & 0xFFFFFFFFu),         // global_addr[31:0]
                     (unsigned)((ga >> 32) & 0x1FFFFFFu)   // global_addr[56:32]
                         | (2u << 30) };                   // type=2 ("image")
        i32x8 g1 = { 0x00010000,          // data_size=1 (2 bytes) @ [17:16]
                     0x00200000,          // tensor_dim0=32 @ [79:48] (low half)
                     (int)(2048u << 16),  // tensor_dim1=2048 @ [111:80]
                     (int)(32u << 16),    // tile_dim0=32 @ [127:112]
                     2048,                // tile_dim1=2048 @ [143:128]
                     32,                  // tensor_dim0_stride=32 @ [207:160]
                     0, 0 };
        i32x4 gz = { 0, 0, 0, 0 };
#if defined(__clang_major__) && __clang_major__ >= 23
        i32x8 gz8 = { 0, 0, 0, 0, 0, 0, 0, 0 };
        __builtin_amdgcn_tensor_load_to_lds(g0, g1, gz, gz, gz8, 0);
#else
        __builtin_amdgcn_tensor_load_to_lds(g0, g1, gz, gz, 0);
#endif
    }

    // Per-row features: c = (x)_{i<4} (cos h_i, -i sin h_i) => 32 reals.
    if (tid < 128) {
        int b = blockIdx.x * 128 + tid;
        float cs[4], sn[4];
#pragma unroll
        for (int i = 0; i < 4; ++i) {
            float h = 0.5f * x[b * 4 + i];
            cs[i] = cosf(h);
            sn[i] = sinf(h);
        }
#pragma unroll
        for (int k = 0; k < 16; ++k) {
            float r = 1.0f;
#pragma unroll
            for (int i = 0; i < 4; ++i)
                r *= ((k >> (3 - i)) & 1) ? sn[i] : cs[i];
            int m = __popc(k) & 3;
            float cr = (m == 0) ? r : ((m == 2) ? -r : 0.0f);
            float ci = (m == 1) ? -r : ((m == 3) ? r : 0.0f);
            feat[tid * 32 + k]      = (_Float16)cr;
            feat[tid * 32 + 16 + k] = (_Float16)ci;
        }
    }

    if (wave == 0) __builtin_amdgcn_s_wait_tensorcnt(0);
    __syncthreads();   // publishes TDM data + features to all waves

    // A operand: row M=nrow of this wave's 16x32 feature tile.
    // A layout (matches B): lanes<16 hold K=0..15, lanes>=16 K=16..31
    int hsel = (lane & 16) ? 16 : 0;
    const _Float16* frow = feat + (wave * 16 + nrow) * 32;
    v8h alo = *(const v8h*)(frow + hsel);
    v8h ahi = *(const v8h*)(frow + hsel + 8);
    v16h a;
#pragma unroll
    for (int e = 0; e < 8; ++e) { a[e] = alo[e]; a[e + 8] = ahi[e]; }

    float acc[10][8];
#pragma unroll
    for (int w = 0; w < 10; ++w)
#pragma unroll
        for (int v = 0; v < 8; ++v) acc[w][v] = 0.0f;

    int boff = (lane & 16) ? 16 : 0;   // B layout: lanes<16 K=0..15, >=16 K=16..31
    for (int t = 0; t < 64; ++t) {
        v16h bp = *(const v16h*)(PQ + (t * 16 + nrow) * 32 + boff);
        v16h bq = *(const v16h*)(PQ + (QDIM + t * 16 + nrow) * 32 + boff);

        v8f cre = {};
        v8f cim = {};
        cre = __builtin_amdgcn_wmma_f32_16x16x32_f16(false, a, false, bp,
                                                     (short)0, cre, false, false);
        cim = __builtin_amdgcn_wmma_f32_16x16x32_f16(false, a, false, bq,
                                                     (short)0, cim, false, false);

        int icol = (t << 4) | nrow;
        float sgn[10];
#pragma unroll
        for (int w = 0; w < 10; ++w)
            sgn[w] = ((icol >> (9 - w)) & 1) ? -1.0f : 1.0f;

#pragma unroll
        for (int v = 0; v < 8; ++v) {
            float p = fmaf(cre[v], cre[v], cim[v] * cim[v]);
#pragma unroll
            for (int w = 0; w < 10; ++w)
                acc[w][v] = fmaf(sgn[w], p, acc[w][v]);
        }
    }

    // Reduce the 16 state-columns spread across each 16-lane half-group.
#pragma unroll
    for (int v = 0; v < 8; ++v) {
#pragma unroll
        for (int w = 0; w < 10; ++w) {
            float s = acc[w][v];
            s += __shfl_xor(s, 1, 32);
            s += __shfl_xor(s, 2, 32);
            s += __shfl_xor(s, 4, 32);
            s += __shfl_xor(s, 8, 32);
            if (nrow == 0) {
                int row = blockIdx.x * 128 + wave * 16 + v + ((lane & 16) ? 8 : 0);
                out[row * 10 + w] = s;
            }
        }
    }
}

// ---------------------------------------------------------------------------
// Two launches total (graph-capture safe).  ws: P||Q = 128KB at offset 0.
// ---------------------------------------------------------------------------
extern "C" void kernel_launch(void* const* d_in, const int* in_sizes, int n_in,
                              void* d_out, int out_size, void* d_ws, size_t ws_size,
                              hipStream_t stream) {
    (void)in_sizes; (void)n_in; (void)out_size; (void)ws_size;
    const float* x     = (const float*)d_in[0];
    const float* theta = (const float*)d_in[1];
    float* out = (float*)d_out;
    _Float16* PQ = (_Float16*)d_ws;

    k_precompute<<<1, 1024, 131072, stream>>>(theta, PQ);
    k_eval<<<128, 256, 139264, stream>>>(x, PQ, out);
}
